// BeliefUpdater_3659312136484
// MI455X (gfx1250) — matrix operand order repaired and checked by reading, not compiled, not run on hardware
//
#include <hip/hip_runtime.h>

// ---------------------------------------------------------------------------
// BeliefUpdater for MI455X (gfx1250, wave32, WMMA).
//  * Weights pre-transposed to bf16 [N][K] once per launch.
//  * ctx_in broadcast structure factored out: big GEMMs are K=1024 slot GEMMs.
//  * v_wmma_f32_16x16x32_bf16 tiled GEMM: 128x64 block, 8 waves, 32x32/wave.
//  * B (weight) tiles staged global->LDS with GLOBAL_LOAD_ASYNC_TO_LDS_B128
//    (double-buffered, ASYNCcnt-sequenced); A tiles prefetched through VGPRs
//    (fp32->bf16 convert on the fly, or raw bf16 for the hidden activations).
//  * Hidden activations kept in bf16 (halves h1 HBM traffic).
//  * Score heads (W2 [H,1]) fused: GELU + dot(W2) + cross-lane reduce ->
//    per-chunk partials, summed by a deterministic reduce kernel (no atomics).
// ---------------------------------------------------------------------------

typedef __bf16 bf16_t;
typedef bf16_t v8bf  __attribute__((ext_vector_type(8)));
typedef bf16_t v16bf __attribute__((ext_vector_type(16)));
typedef float  v8f   __attribute__((ext_vector_type(8)));
typedef float  v4f   __attribute__((ext_vector_type(4)));

#define TILE_M 128
#define TILE_N 64
#define TILE_K 32
#define LDSA_STRIDE 40   // 32 + 8 pad (bf16 elems) -> conflict-free b128 reads
#define LDSB_STRIDE 40

__device__ __forceinline__ float gelu_exact(float x) {
    return 0.5f * x * (1.0f + erff(x * 0.70710678118654752440f));
}

// Async global->LDS 16B copy per lane (CDNA5 TDM-adjacent path, ASYNCcnt).
__device__ __forceinline__ void async_ld_b128(unsigned lds_off, const void* gptr) {
    asm volatile("global_load_async_to_lds_b128 %0, %1, off"
                 :: "v"(lds_off), "v"(gptr) : "memory");
}
__device__ __forceinline__ void wait_async_le1() {
    asm volatile("s_wait_asynccnt 0x1" ::: "memory");
}
__device__ __forceinline__ void wait_async_le0() {
    asm volatile("s_wait_asynccnt 0x0" ::: "memory");
}

// --------------------------- transpose + convert ---------------------------
// src fp32 [R][C] -> dst bf16 [C][R]
__global__ __launch_bounds__(256) void transpose_to_bf16(
    const float* __restrict__ src, bf16_t* __restrict__ dst, int R, int C)
{
    __shared__ float tile[32][33];
    const int c0 = blockIdx.x * 32, r0 = blockIdx.y * 32;
    const int tx = threadIdx.x & 31, ty = threadIdx.x >> 5;   // 32x8
    for (int i = 0; i < 32; i += 8)
        tile[ty + i][tx] = src[(size_t)(r0 + ty + i) * C + c0 + tx];
    __syncthreads();
    for (int i = 0; i < 32; i += 8)
        dst[(size_t)(c0 + ty + i) * R + r0 + tx] = (bf16_t)tile[tx][ty + i];
}

// ------------------------------- WMMA GEMM ---------------------------------
// C[M,N] = epilogue( A[M,K] @ Bt[N,K]^T )
// A is fp32 (ABF16=false) or bf16 (ABF16=true).
// epilogue: (+bias[n]) (+ctx[row>>gshift][n]) (+residual[row][n]) (gelu)
// OUTBF16: store bf16 (out reinterpreted).
// FUSED:   out[row, colchunk] = sum_n gelu(...)*w2[n]  per 32-col chunk.
template<bool ABF16, bool OUTBF16, bool FUSED>
__global__ __launch_bounds__(256)
void wmma_gemm_bf16(const void* __restrict__ A, int lda,
                    const bf16_t* __restrict__ Bt, int ldb,
                    const float* __restrict__ bias,
                    const float* __restrict__ ctx, int ctxld, int gshift,
                    const float* __restrict__ residual, int ldres,
                    int do_gelu,
                    float* __restrict__ out, int ldout,
                    const float* __restrict__ w2,
                    int K)
{
    __shared__ bf16_t sA[TILE_M * LDSA_STRIDE];
    __shared__ bf16_t sB[2][TILE_N * LDSB_STRIDE];   // double-buffered (async)

    const int tid  = threadIdx.x;
    const int m0   = blockIdx.y * TILE_M;
    const int n0   = blockIdx.x * TILE_N;
    const int wave = tid >> 5;
    const int lane = tid & 31;
    const int wm   = wave >> 1;          // 0..3 -> 32-row strip
    const int wn   = wave & 1;           // 0..1 -> 32-col strip
    const int rbase = wm * 32;
    const int cbase = wn * 32;
    const int lm   = lane & 15;
    const int hi   = lane >> 4;

    v8f acc[2][2];
    for (int mi = 0; mi < 2; ++mi)
        for (int ni = 0; ni < 2; ++ni)
            for (int v = 0; v < 8; ++v) acc[mi][ni][v] = 0.0f;

    // staging: A 128x32 (2 thr/row); B 64x32 (4 thr/row, 16B async per thr)
    const int am = tid >> 1, ak = (tid & 1) * 16;
    const int bn = tid >> 2, bk = (tid & 3) * 8;
    const float*  aPtrF = nullptr;
    const bf16_t* aPtrH = nullptr;
    if constexpr (ABF16) aPtrH = (const bf16_t*)A + (size_t)(m0 + am) * lda + ak;
    else                 aPtrF = (const float*)A  + (size_t)(m0 + am) * lda + ak;
    const bf16_t* bSrc = Bt + (size_t)(n0 + bn) * ldb + bk;
    const unsigned sBoff0 = (unsigned)(uintptr_t)&sB[0][bn * LDSB_STRIDE + bk];
    const unsigned sBoff1 = (unsigned)(uintptr_t)&sB[1][bn * LDSB_STRIDE + bk];

    v4f f0, f1, f2, f3; v8bf h0, h1r;
    auto load_tile = [&](int k0) {
        if constexpr (ABF16) {
            h0  = *(const v8bf*)(aPtrH + k0);
            h1r = *(const v8bf*)(aPtrH + k0 + 8);
        } else {
            f0 = *(const v4f*)(aPtrF + k0);
            f1 = *(const v4f*)(aPtrF + k0 + 4);
            f2 = *(const v4f*)(aPtrF + k0 + 8);
            f3 = *(const v4f*)(aPtrF + k0 + 12);
        }
    };
    auto store_tile = [&]() {
        if constexpr (ABF16) {
            *(v8bf*)&sA[am * LDSA_STRIDE + ak]     = h0;
            *(v8bf*)&sA[am * LDSA_STRIDE + ak + 8] = h1r;
        } else {
            v8bf c0, c1;
            for (int j = 0; j < 4; ++j) {
                c0[j] = (bf16_t)f0[j]; c0[4 + j] = (bf16_t)f1[j];
                c1[j] = (bf16_t)f2[j]; c1[4 + j] = (bf16_t)f3[j];
            }
            *(v8bf*)&sA[am * LDSA_STRIDE + ak]     = c0;
            *(v8bf*)&sA[am * LDSA_STRIDE + ak + 8] = c1;
        }
    };

    async_ld_b128(sBoff0, (const void*)bSrc);      // B tile 0 -> sB[0]
    load_tile(0);
    int par = 0;
    for (int k0 = 0; k0 < K; k0 += TILE_K, par ^= 1) {
        store_tile();                              // A regs -> LDS
        const bool more = (k0 + TILE_K < K);
        if (more) {
            // next B tile into the other buffer (its readers passed the
            // barrier one iteration ago), overlapped with this tile's math
            async_ld_b128(par ? sBoff0 : sBoff1,
                          (const void*)(bSrc + k0 + TILE_K));
            load_tile(k0 + TILE_K);
            wait_async_le1();                      // oldest (current B) done
        } else {
            wait_async_le0();
        }
        __syncthreads();

        // A frag: lane row = lm, K = hi*8 + {0..7, 16..23}
        const int kbA = hi * 8;
        // B frag: lane col = lm, K = hi*8 + {0..7, 16..23}
        const int kbB = hi * 8;
        v16bf af[2], bfr[2];
        for (int mi = 0; mi < 2; ++mi) {
            const bf16_t* pa = &sA[(rbase + mi * 16 + lm) * LDSA_STRIDE + kbA];
            v8bf lo = *(const v8bf*)pa;
            v8bf hh = *(const v8bf*)(pa + 16);
            af[mi] = __builtin_shufflevector(lo, hh,
                0,1,2,3,4,5,6,7,8,9,10,11,12,13,14,15);
        }
        for (int ni = 0; ni < 2; ++ni) {
            const bf16_t* pb = &sB[par][(cbase + ni * 16 + lm) * LDSB_STRIDE + kbB];
            v8bf lo = *(const v8bf*)pb;
            v8bf hh = *(const v8bf*)(pb + 16);
            bfr[ni] = __builtin_shufflevector(lo, hh,
                0,1,2,3,4,5,6,7,8,9,10,11,12,13,14,15);
        }
        for (int mi = 0; mi < 2; ++mi)
            for (int ni = 0; ni < 2; ++ni)
                acc[mi][ni] = __builtin_amdgcn_wmma_f32_16x16x32_bf16(
                    false, af[mi], false, bfr[ni], (short)0, acc[mi][ni],
                    false, false);
        __syncthreads();
    }

    // epilogue. C layout: VGPR v, lane l -> M = v + 8*(l>>4), N = l&15
    if (FUSED) {
        for (int mi = 0; mi < 2; ++mi) {
            float rs[8];
            for (int v = 0; v < 8; ++v) rs[v] = 0.0f;
            for (int ni = 0; ni < 2; ++ni) {
                const int col  = n0 + cbase + ni * 16 + lm;
                const float bcol = bias ? bias[col] : 0.0f;
                const float wcol = w2[col];
                for (int v = 0; v < 8; ++v) {
                    const int row = m0 + rbase + mi * 16 + hi * 8 + v;
                    float val = acc[mi][ni][v] + bcol;
                    if (ctx) val += ctx[(size_t)(row >> gshift) * ctxld + col];
                    if (do_gelu) val = gelu_exact(val);
                    rs[v] += val * wcol;
                }
            }
            const int chunk = blockIdx.x * 2 + wn;   // 32-col chunk id
            for (int v = 0; v < 8; ++v) {
                float p = rs[v];
                p += __shfl_xor(p, 1);
                p += __shfl_xor(p, 2);
                p += __shfl_xor(p, 4);
                p += __shfl_xor(p, 8);
                if (lm == 0) {
                    const int row = m0 + rbase + mi * 16 + hi * 8 + v;
                    out[(size_t)row * ldout + chunk] = p;   // partial, no atomics
                }
            }
        }
    } else {
        for (int mi = 0; mi < 2; ++mi) {
            for (int ni = 0; ni < 2; ++ni) {
                const int col  = n0 + cbase + ni * 16 + lm;
                const float bcol = bias ? bias[col] : 0.0f;
                for (int v = 0; v < 8; ++v) {
                    const int row = m0 + rbase + mi * 16 + hi * 8 + v;
                    float val = acc[mi][ni][v] + bcol;
                    if (ctx)      val += ctx[(size_t)(row >> gshift) * ctxld + col];
                    if (residual) val += residual[(size_t)row * ldres + col];
                    if (do_gelu)  val = gelu_exact(val);
                    if (OUTBF16)
                        ((bf16_t*)out)[(size_t)row * ldout + col] = (bf16_t)val;
                    else
                        out[(size_t)row * ldout + col] = val;
                }
            }
        }
    }
}

// --------------------------- small helper kernels --------------------------
// dst[256][3*1024] = [a | b | c]
__global__ __launch_bounds__(256) void concat3_kernel(
    const float* __restrict__ a, const float* __restrict__ b,
    const float* __restrict__ c, float* __restrict__ dst)
{
    const int i = blockIdx.x * 256 + threadIdx.x;     // 256*3072 total
    const int row = i / 3072;
    const int col = i - row * 3072;
    float v;
    if (col < 1024)       v = a[row * 1024 + col];
    else if (col < 2048)  v = b[row * 1024 + col - 1024];
    else                  v = c[row * 1024 + col - 2048];
    dst[i] = v;
}

// dst[i] = (base?base[i]:0) + b2[0] + sum_p partial[i*np+p]   (fixed order)
__global__ __launch_bounds__(256) void score_reduce_kernel(
    const float* __restrict__ partial, int np,
    const float* __restrict__ base, const float* __restrict__ b2,
    float* __restrict__ dst, int n)
{
    const int i = blockIdx.x * 256 + threadIdx.x;
    if (i >= n) return;
    float s = (base ? base[i] : 0.0f) + b2[0];
    for (int p = 0; p < np; ++p) s += partial[(size_t)i * np + p];
    dst[i] = s;
}

// pooled[b,d] = sum_k softmax(scores[b,:])[k] * slots[b,k,d]
__global__ __launch_bounds__(256) void softmax_pool_kernel(
    const float* __restrict__ slots, const float* __restrict__ scores,
    float* __restrict__ pooled)
{
    __shared__ float w[64];
    const int b = blockIdx.x, tid = threadIdx.x;
    if (tid == 0) {
        float mx = -3.0e38f;
        for (int k = 0; k < 64; ++k) mx = fmaxf(mx, scores[b * 64 + k]);
        float s = 0.0f;
        for (int k = 0; k < 64; ++k) { float e = expf(scores[b*64+k] - mx); w[k] = e; s += e; }
        const float inv = 1.0f / s;
        for (int k = 0; k < 64; ++k) w[k] *= inv;
    }
    __syncthreads();
    for (int c = tid; c < 1024; c += 256) {
        float a = 0.0f;
        for (int k = 0; k < 64; ++k)
            a += w[k] * slots[((size_t)b * 64 + k) * 1024 + c];
        pooled[(size_t)b * 1024 + c] = a;
    }
}

// top-16 of [archive_scores(16) | a_scores(64)]; gather slots. Ties -> lower idx.
__global__ __launch_bounds__(256) void topk_kernel(
    const float* __restrict__ arch_slots, const float* __restrict__ arch_scores,
    const float* __restrict__ a_slots,    const float* __restrict__ a_scores,
    float* __restrict__ top_slots, float* __restrict__ top_vals)
{
    __shared__ float sc[80];
    __shared__ int   sidx[16];
    __shared__ float sval[16];
    const int b = blockIdx.x, tid = threadIdx.x;
    if (tid < 16)                sc[tid] = arch_scores[b * 16 + tid];
    else if (tid < 80)           sc[tid] = a_scores[b * 64 + (tid - 16)];
    __syncthreads();
    if (tid == 0) {
        for (int i = 0; i < 16; ++i) {
            float best = -3.0e38f; int bj = 0;
            for (int j = 0; j < 80; ++j)
                if (sc[j] > best) { best = sc[j]; bj = j; }
            sval[i] = best; sidx[i] = bj; sc[bj] = -3.4e38f;
        }
    }
    __syncthreads();
    if (tid < 16) top_vals[b * 16 + tid] = sval[tid];
    for (int i = 0; i < 16; ++i) {
        const int idx = sidx[i];
        const float* src = (idx < 16)
            ? (arch_slots + ((size_t)b * 16 + idx) * 1024)
            : (a_slots    + ((size_t)b * 64 + (idx - 16)) * 1024);
        float* dst = top_slots + ((size_t)b * 16 + i) * 1024;
        for (int c = tid; c < 1024; c += 256) dst[c] = src[c];
    }
}

// ------------------------------- host side ---------------------------------
extern "C" void kernel_launch(void* const* d_in, const int* in_sizes, int n_in,
                              void* d_out_v, int out_size, void* d_ws, size_t ws_size,
                              hipStream_t stream)
{
    (void)in_sizes; (void)n_in; (void)out_size; (void)ws_size;

    const float* in_ctx       = (const float*)d_in[0];
    const float* in_a_slots   = (const float*)d_in[1];
    const float* in_a_scores  = (const float*)d_in[2];
    const float* in_r_slots   = (const float*)d_in[3];
    const float* in_r_scores  = (const float*)d_in[4];
    const float* in_probe_emb = (const float*)d_in[5];
    const float* in_obs_emb   = (const float*)d_in[6];
    const float* in_arch_slots  = (const float*)d_in[7];
    const float* in_arch_scores = (const float*)d_in[8];
    const float* Wp = (const float*)d_in[9],  *bp = (const float*)d_in[10];
    const float* Wo = (const float*)d_in[11], *bo = (const float*)d_in[12];
    const float* ad_W1 = (const float*)d_in[13], *ad_b1 = (const float*)d_in[14];
    const float* ad_W2 = (const float*)d_in[15], *ad_b2 = (const float*)d_in[16];
    const float* rd_W1 = (const float*)d_in[17], *rd_b1 = (const float*)d_in[18];
    const float* rd_W2 = (const float*)d_in[19], *rd_b2 = (const float*)d_in[20];
    const float* au_W1 = (const float*)d_in[21], *au_b1 = (const float*)d_in[22];
    const float* au_W2 = (const float*)d_in[23], *au_b2 = (const float*)d_in[24];
    const float* ru_W1 = (const float*)d_in[25], *ru_b1 = (const float*)d_in[26];
    const float* ru_W2 = (const float*)d_in[27], *ru_b2 = (const float*)d_in[28];
    const float* s_W1  = (const float*)d_in[29], *s_b1  = (const float*)d_in[30];
    const float* s_W2  = (const float*)d_in[31], *s_b2  = (const float*)d_in[32];
    const float* st_W1 = (const float*)d_in[33], *st_b1 = (const float*)d_in[34];
    const float* st_W2 = (const float*)d_in[35], *st_b2 = (const float*)d_in[36];

    float* out = (float*)d_out_v;
    float* out_a_slots  = out;                 // 256*64*1024
    float* out_a_scores = out + 16777216;      // 256*64
    float* out_r_slots  = out + 16793600;      // 256*64*1024
    float* out_r_scores = out + 33570816;      // 256*64
    float* out_top_slots= out + 33587200;      // 256*16*1024
    float* out_top_vals = out + 37781504;      // 256*16
    float* out_surprise = out + 37785600;      // 256
    float* out_stag     = out + 37785856;      // 256

    uintptr_t p = (uintptr_t)d_ws;
    auto take = [&](size_t bytes) -> void* {
        void* r = (void*)p; p += (bytes + 255) & ~(size_t)255; return r;
    };
    bf16_t* WpT   = (bf16_t*)take((size_t)1024*1024*2);
    bf16_t* WoT   = (bf16_t*)take((size_t)1024*1024*2);
    bf16_t* auW1T = (bf16_t*)take((size_t)4096*1024*2);
    bf16_t* ruW1T = (bf16_t*)take((size_t)4096*1024*2);
    bf16_t* adW1T = (bf16_t*)take((size_t)4096*2048*2);
    bf16_t* rdW1T = (bf16_t*)take((size_t)4096*2048*2);
    bf16_t* auW2T = (bf16_t*)take((size_t)1024*1024*2);
    bf16_t* ruW2T = (bf16_t*)take((size_t)1024*1024*2);
    bf16_t* sW1T  = (bf16_t*)take((size_t)3072*1024*2);
    bf16_t* stW1T = (bf16_t*)take((size_t)3072*1024*2);
    float* probe   = (float*)take((size_t)256*1024*4);
    float* obs     = (float*)take((size_t)256*1024*4);
    float* X3a     = (float*)take((size_t)256*3072*4);
    float* X3s     = (float*)take((size_t)256*3072*4);
    float* X3st    = (float*)take((size_t)256*3072*4);
    float* ctx_au  = (float*)take((size_t)256*1024*4);
    float* ctx_ru  = (float*)take((size_t)256*1024*4);
    float* ctx_ad  = (float*)take((size_t)256*2048*4);
    float* ctx_rd  = (float*)take((size_t)256*2048*4);
    float* pooled_a= (float*)take((size_t)256*1024*4);
    float* pooled_r= (float*)take((size_t)256*1024*4);
    bf16_t* h1b    = (bf16_t*)take((size_t)16384*1024*2);   // hidden, bf16
    float* part    = (float*)take((size_t)16384*64*4);

    const dim3 blk(256);

    // ---- Phase 0: transpose-convert weights to bf16 [N][K]
    transpose_to_bf16<<<dim3(32,  32), blk, 0, stream>>>(Wp,    WpT,   1024, 1024);
    transpose_to_bf16<<<dim3(32,  32), blk, 0, stream>>>(Wo,    WoT,   1024, 1024);
    transpose_to_bf16<<<dim3(32, 128), blk, 0, stream>>>(au_W1, auW1T, 4096, 1024);
    transpose_to_bf16<<<dim3(32, 128), blk, 0, stream>>>(ru_W1, ruW1T, 4096, 1024);
    transpose_to_bf16<<<dim3(64, 128), blk, 0, stream>>>(ad_W1, adW1T, 4096, 2048);
    transpose_to_bf16<<<dim3(64, 128), blk, 0, stream>>>(rd_W1, rdW1T, 4096, 2048);
    transpose_to_bf16<<<dim3(32,  32), blk, 0, stream>>>(au_W2, auW2T, 1024, 1024);
    transpose_to_bf16<<<dim3(32,  32), blk, 0, stream>>>(ru_W2, ruW2T, 1024, 1024);
    transpose_to_bf16<<<dim3(32,  96), blk, 0, stream>>>(s_W1,  sW1T,  3072, 1024);
    transpose_to_bf16<<<dim3(32,  96), blk, 0, stream>>>(st_W1, stW1T, 3072, 1024);

    // ---- Phase 1: probe / obs projections  [256,1024]
    wmma_gemm_bf16<false,false,false><<<dim3(16, 2), blk, 0, stream>>>(
        in_probe_emb, 1024, WpT, 1024, bp, nullptr, 0, 0, nullptr, 0, 0,
        probe, 1024, nullptr, 1024);
    wmma_gemm_bf16<false,false,false><<<dim3(16, 2), blk, 0, stream>>>(
        in_obs_emb, 1024, WoT, 1024, bo, nullptr, 0, 0, nullptr, 0, 0,
        obs, 1024, nullptr, 1024);

    // ---- Phase 2: concat buffers
    concat3_kernel<<<3072, blk, 0, stream>>>(probe, obs, in_ctx, X3a);   // ctx_in tail order
    concat3_kernel<<<3072, blk, 0, stream>>>(in_ctx, probe, obs, X3s);   // surprise input

    // ---- Phase 3: per-batch ctx parts (W1 rows D..4D-1 == Wt cols D..4D-1)
    wmma_gemm_bf16<false,false,false><<<dim3(16, 2), blk, 0, stream>>>(
        X3a, 3072, auW1T + 1024, 4096, au_b1, nullptr, 0, 0, nullptr, 0, 0,
        ctx_au, 1024, nullptr, 3072);
    wmma_gemm_bf16<false,false,false><<<dim3(32, 2), blk, 0, stream>>>(
        X3a, 3072, adW1T + 1024, 4096, ad_b1, nullptr, 0, 0, nullptr, 0, 0,
        ctx_ad, 2048, nullptr, 3072);
    wmma_gemm_bf16<false,false,false><<<dim3(16, 2), blk, 0, stream>>>(
        X3a, 3072, ruW1T + 1024, 4096, ru_b1, nullptr, 0, 0, nullptr, 0, 0,
        ctx_ru, 1024, nullptr, 3072);
    wmma_gemm_bf16<false,false,false><<<dim3(32, 2), blk, 0, stream>>>(
        X3a, 3072, rdW1T + 1024, 4096, rd_b1, nullptr, 0, 0, nullptr, 0, 0,
        ctx_rd, 2048, nullptr, 3072);

    // ---- Phase 4: approach slots + scores  (rows = B*K = 16384)
    wmma_gemm_bf16<false,true,false><<<dim3(16, 128), blk, 0, stream>>>(  // hidden (bf16 out)
        in_a_slots, 1024, auW1T, 4096, nullptr, ctx_au, 1024, 6,
        nullptr, 0, 1, (float*)h1b, 1024, nullptr, 1024);
    wmma_gemm_bf16<true,false,false><<<dim3(16, 128), blk, 0, stream>>>(  // out + residual
        h1b, 1024, auW2T, 1024, au_b2, nullptr, 0, 0,
        in_a_slots, 1024, 0, out_a_slots, 1024, nullptr, 1024);
    wmma_gemm_bf16<false,false,true><<<dim3(32, 128), blk, 0, stream>>>(  // fused score
        in_a_slots, 1024, adW1T, 4096, nullptr, ctx_ad, 2048, 6,
        nullptr, 0, 1, part, 64, ad_W2, 1024);
    score_reduce_kernel<<<64, blk, 0, stream>>>(part, 64, in_a_scores, ad_b2,
                                                out_a_scores, 16384);

    // ---- Phase 5: rules (reuse h1b, part)
    wmma_gemm_bf16<false,true,false><<<dim3(16, 128), blk, 0, stream>>>(
        in_r_slots, 1024, ruW1T, 4096, nullptr, ctx_ru, 1024, 6,
        nullptr, 0, 1, (float*)h1b, 1024, nullptr, 1024);
    wmma_gemm_bf16<true,false,false><<<dim3(16, 128), blk, 0, stream>>>(
        h1b, 1024, ruW2T, 1024, ru_b2, nullptr, 0, 0,
        in_r_slots, 1024, 0, out_r_slots, 1024, nullptr, 1024);
    wmma_gemm_bf16<false,false,true><<<dim3(32, 128), blk, 0, stream>>>(
        in_r_slots, 1024, rdW1T, 4096, nullptr, ctx_rd, 2048, 6,
        nullptr, 0, 1, part, 64, rd_W2, 1024);
    score_reduce_kernel<<<64, blk, 0, stream>>>(part, 64, in_r_scores, rd_b2,
                                                out_r_scores, 16384);

    // ---- Phase 6: pooling, surprise, stagnation
    softmax_pool_kernel<<<256, blk, 0, stream>>>(out_a_slots, out_a_scores, pooled_a);
    softmax_pool_kernel<<<256, blk, 0, stream>>>(out_r_slots, out_r_scores, pooled_r);

    wmma_gemm_bf16<false,false,true><<<dim3(16, 2), blk, 0, stream>>>(    // surprise
        X3s, 3072, sW1T, 3072, s_b1, nullptr, 0, 0,
        nullptr, 0, 1, part, 32, s_W2, 3072);
    score_reduce_kernel<<<1, blk, 0, stream>>>(part, 32, nullptr, s_b2,
                                               out_surprise, 256);

    concat3_kernel<<<3072, blk, 0, stream>>>(in_ctx, pooled_a, pooled_r, X3st);
    wmma_gemm_bf16<false,false,true><<<dim3(16, 2), blk, 0, stream>>>(    // stagnation
        X3st, 3072, stW1T, 3072, st_b1, nullptr, 0, 0,
        nullptr, 0, 1, part, 32, st_W2, 3072);
    score_reduce_kernel<<<1, blk, 0, stream>>>(part, 32, nullptr, st_b2,
                                               out_stag, 256);

    // ---- Phase 7: top-k
    topk_kernel<<<256, blk, 0, stream>>>(in_arch_slots, in_arch_scores,
                                         out_a_slots, out_a_scores,
                                         out_top_slots, out_top_vals);
}
